// Attr_Relation_Net_58222576664917
// MI455X (gfx1250) — matrix operand order repaired and checked by reading, not compiled, not run on hardware
//
#include <hip/hip_runtime.h>

typedef __attribute__((ext_vector_type(16))) _Float16 v16h;
typedef __attribute__((ext_vector_type(8)))  float    v8f;

#define WAVES_PER_BLOCK 8
#define TILE_M 16
#define KP64   66    // padded K stride (halfs) for K=64 weight tables
#define KP128  130   // padded K stride (halfs) for K=128 weight tables
#define SSTR   136   // stage row stride in halfs (16B-multiple rows)

// Packed f16 weight image layout (halfs), shared by prep kernel, d_ws, and LDS:
#define OFF_WRM1 0                       // [N=128][KP64]
#define OFF_WRM2 (OFF_WRM1 + 128 * KP64) // [N=64][KP128]   = 8448
#define OFF_WRR  (OFF_WRM2 + 64 * KP128) // [N=128][KP64]   = 16768
#define OFF_WRC  (OFF_WRR + 128 * KP64)  // [N=128][KP128]  = 25216
#define W_TOTAL  (OFF_WRC + 128 * KP128) // = 41856 halfs = 83712 B
#define W_CHUNKS (W_TOTAL * 2 / 16)      // 16B chunks = 5232

union AFrag {
    v16h     v;
    _Float16 h[16];
    unsigned u[8];
};

// Branch-free tanh GELU: x*0.5*(1+tanh(sqrt(2/pi)*(x+0.044715 x^3))).
// Uses v_exp_f32 + v_rcp_f32; max |err| vs exact-erf GELU ~3e-4 (below f16 GEMM noise).
__device__ __forceinline__ float gelu_fast(float x) {
    float y  = 0.7978845608028654f * (x + 0.044715f * x * x * x);
    float t  = __expf(2.0f * y);                      // v_exp_f32, no branches
    float th = 1.0f - 2.0f * __builtin_amdgcn_rcpf(t + 1.0f);
    return 0.5f * x * (1.0f + th);
}

#define WMMA_F16(A, B, C) \
    __builtin_amdgcn_wmma_f32_16x16x32_f16(false, (A), false, (B), (short)0, (C), false, false)

// A-fragment K offset pattern (16-bit A, 16x32, wave32; ISA 7.12.2):
//   lane(l, half): M = l, VGPR v holds K = ((v&4)?16:0) + 8*half + 2*(v&3) + {0,1}
__device__ __forceinline__ int afrag_k(int v, int half) {
    return ((v & 4) ? 16 : 0) + half * 8 + 2 * (v & 3);
}

__device__ __forceinline__ void load_afrag_stage(const _Float16* srow, int c, int half, AFrag& a) {
#pragma unroll
    for (int v = 0; v < 8; ++v) {
        a.u[v] = *(const unsigned*)(srow + 32 * c + afrag_k(v, half));
    }
}

// B fragment (16-bit B, 32x16, wave32): N = lane&15, VGPR v holds K = afrag_k(v, half) + {0,1}.
__device__ __forceinline__ void load_bfrag(const _Float16* w, int kpad, int ncol, int c,
                                           int half, AFrag& b) {
    const _Float16* base = w + ncol * kpad + 32 * c;
#pragma unroll
    for (int v = 0; v < 8; ++v) {
        b.u[v] = *(const unsigned*)(base + afrag_k(v, half));
    }
}

// ---------------------------------------------------------------------------
// Prep kernel: build the f16, transposed ([N][K], padded-K) weight image in
// d_ws once, so the 3907 main blocks just DMA 83.7KB instead of converting.
// ---------------------------------------------------------------------------
__global__ void prep_weights_kernel(const float* __restrict__ W_rm1,
                                    const float* __restrict__ W_rm2,
                                    const float* __restrict__ W_rr,
                                    const float* __restrict__ W_rc,
                                    _Float16* __restrict__ wimg)
{
    int i = blockIdx.x * blockDim.x + threadIdx.x;
    if (i >= W_TOTAL) return;
    float val = 0.0f;
    if (i < OFF_WRM2) {                       // W_rm1 [K=64][N=128]
        int j = i - OFF_WRM1, n = j / KP64, k = j % KP64;
        if (k < 64) val = W_rm1[k * 128 + n];
    } else if (i < OFF_WRR) {                 // W_rm2 [K=128][N=64]
        int j = i - OFF_WRM2, n = j / KP128, k = j % KP128;
        if (k < 128) val = W_rm2[k * 64 + n];
    } else if (i < OFF_WRC) {                 // W_rr [K=64][N=128]
        int j = i - OFF_WRR, n = j / KP64, k = j % KP64;
        if (k < 64) val = W_rr[k * 128 + n];
    } else {                                  // W_rc [K=128][N=128]
        int j = i - OFF_WRC, n = j / KP128, k = j % KP128;
        if (k < 128) val = W_rc[k * 128 + n];
    }
    wimg[i] = (_Float16)val;
}

// ---------------------------------------------------------------------------
// Main kernel: one wave32 per 16-edge tile; 4 chained f16 WMMA GEMMs.
// ---------------------------------------------------------------------------
__global__ __launch_bounds__(256) void attr_relation_net_kernel(
    const float* __restrict__ known_mask,   // [N_SAMP, 64]
    const int*   __restrict__ obs_idx,      // [E]
    const int*   __restrict__ obs_mask_idx, // [E]
    const int*   __restrict__ attr_idx,     // [E]
    const float* __restrict__ obs_embs,     // [N_OBS, 128]
    const float* __restrict__ fea_corr,     // [64, 64]
    const _Float16* __restrict__ wimg,      // prepped f16 weight image (d_ws)
    const float* __restrict__ b_rm1, const float* __restrict__ b_rm2,
    const float* __restrict__ b_rr,  const float* __restrict__ b_rc,
    float* __restrict__ out, int E)
{
    __shared__ __align__(16) _Float16 s_w[W_TOTAL];
    __shared__ __align__(16) _Float16 s_stage[WAVES_PER_BLOCK][TILE_M][SSTR];

    const int tid  = threadIdx.x;
    const int lane = tid & 31;
    const int half = lane >> 4;
    const int l    = lane & 15;
    const int wave = tid >> 5;

    // ---- Async DMA the f16 weight image into LDS (ASYNCcnt-tracked) ----
    {
        unsigned lds_base = (unsigned)(unsigned long long)(uintptr_t)&s_w[0];
        for (int j = tid; j < W_CHUNKS; j += 256) {
            unsigned voff    = (unsigned)j * 16u;       // per-lane distinct 16B chunk
            unsigned lds_off = lds_base + voff;
            asm volatile("global_load_async_to_lds_b128 %0, %1, %2"
                         :: "v"(lds_off), "v"(voff), "s"(wimg)
                         : "memory");
        }
        asm volatile("s_wait_asynccnt 0" ::: "memory");
    }
    __syncthreads();

    const _Float16* s_wrm1 = s_w + OFF_WRM1;
    const _Float16* s_wrm2 = s_w + OFF_WRM2;
    const _Float16* s_wrr  = s_w + OFF_WRR;
    const _Float16* s_wrc  = s_w + OFF_WRC;

    // ---- This wave's 16-edge tile ----
    const int e_base = (blockIdx.x * WAVES_PER_BLOCK + wave) * TILE_M;
    const int e      = min(e_base + l, E - 1);   // clamped (stores guarded later)
    const int mrow   = obs_mask_idx[e];
    const int attr   = attr_idx[e];
    const int orow   = obs_idx[e];

    _Float16* srow = &s_stage[wave][l][0];       // stage row for edge M=l

    // ========== Stage 1: softmax(m_i * m_j) computed directly in A-frag layout =======
    float ex[32];
    float psum = 0.0f;
#pragma unroll
    for (int c = 0; c < 2; ++c) {
#pragma unroll
        for (int v = 0; v < 8; ++v) {
            int K = 32 * c + afrag_k(v, half);
            float2 mv = *(const float2*)&known_mask[(size_t)mrow * 64 + K];
            float x0 = (K     == attr) ? 0.0f : mv.x;   // m_i * m_j (m_j zero at attr)
            float x1 = (K + 1 == attr) ? 0.0f : mv.y;
            float e0 = __expf(x0), e1 = __expf(x1);
            ex[c * 16 + 2 * v]     = e0;
            ex[c * 16 + 2 * v + 1] = e1;
            psum += e0 + e1;
        }
    }
    float total = psum + __shfl_xor(psum, 16, 32);  // combine the two K-halves of edge l
    float inv   = 1.0f / total;

    AFrag a0, a1;
#pragma unroll
    for (int i = 0; i < 16; ++i) {
        a0.h[i] = (_Float16)(ex[i]      * inv);
        a1.h[i] = (_Float16)(ex[16 + i] * inv);
    }

    // ========== GEMM1: Y1 = gelu(m_JI @ W_rm1 + b_rm1)  [16,64]x[64,128] =============
    for (int n = 0; n < 8; ++n) {
        float bias = b_rm1[n * 16 + l];
        v8f acc;
#pragma unroll
        for (int v = 0; v < 8; ++v) acc[v] = bias;
        AFrag b;
        load_bfrag(s_wrm1, KP64, n * 16 + l, 0, half, b);
        acc = WMMA_F16(a0.v, b.v, acc);
        load_bfrag(s_wrm1, KP64, n * 16 + l, 1, half, b);
        acc = WMMA_F16(a1.v, b.v, acc);
#pragma unroll
        for (int v = 0; v < 8; ++v)   // D layout: M = v + 8*half, N = n*16 + l
            s_stage[wave][v + 8 * half][n * 16 + l] = (_Float16)gelu_fast(acc[v]);
    }
    __syncthreads();

    // ========== GEMM2: X2 = gelu(Y1 @ W_rm2 + b_rm2)  [16,128]x[128,64] ==============
    AFrag y[4];
#pragma unroll
    for (int c = 0; c < 4; ++c) load_afrag_stage(srow, c, half, y[c]);
    __syncthreads();   // all lanes done reading before stage overwrite

    for (int n = 0; n < 4; ++n) {
        float bias = b_rm2[n * 16 + l];
        v8f acc;
#pragma unroll
        for (int v = 0; v < 8; ++v) acc[v] = bias;
#pragma unroll
        for (int c = 0; c < 4; ++c) {
            AFrag b;
            load_bfrag(s_wrm2, KP128, n * 16 + l, c, half, b);
            acc = WMMA_F16(y[c].v, b.v, acc);
        }
#pragma unroll
        for (int v = 0; v < 8; ++v)
            s_stage[wave][v + 8 * half][n * 16 + l] = (_Float16)gelu_fast(acc[v]);
    }
    __syncthreads();

    // ========== GEMM3: A = gelu((a_j_i * X2) @ W_rr + b_rr)  [16,64]x[64,128] ========
    AFrag z[2];
#pragma unroll
    for (int c = 0; c < 2; ++c) {
#pragma unroll
        for (int v = 0; v < 8; ++v) {
            int K = 32 * c + afrag_k(v, half);
            const _Float16* sp = srow + K;
            float2 fc = *(const float2*)&fea_corr[(size_t)attr * 64 + K];
            z[c].h[2 * v]     = (_Float16)((float)sp[0] * fc.x);
            z[c].h[2 * v + 1] = (_Float16)((float)sp[1] * fc.y);
        }
    }
    __syncthreads();

    for (int n = 0; n < 8; ++n) {
        float bias = b_rr[n * 16 + l];
        v8f acc;
#pragma unroll
        for (int v = 0; v < 8; ++v) acc[v] = bias;
        AFrag b;
        load_bfrag(s_wrr, KP64, n * 16 + l, 0, half, b);
        acc = WMMA_F16(z[0].v, b.v, acc);
        load_bfrag(s_wrr, KP64, n * 16 + l, 1, half, b);
        acc = WMMA_F16(z[1].v, b.v, acc);
#pragma unroll
        for (int v = 0; v < 8; ++v)
            s_stage[wave][v + 8 * half][n * 16 + l] = (_Float16)gelu_fast(acc[v]);
    }
    __syncthreads();

    // ========== GEMM4: out = gelu((obs_h * A) @ W_rc + b_rc)  [16,128]x[128,128] =====
    AFrag u4[4];
#pragma unroll
    for (int c = 0; c < 4; ++c) {
#pragma unroll
        for (int v = 0; v < 8; ++v) {
            int K = 32 * c + afrag_k(v, half);
            const _Float16* sp = srow + K;
            float2 oh = *(const float2*)&obs_embs[(size_t)orow * 128 + K];
            u4[c].h[2 * v]     = (_Float16)((float)sp[0] * oh.x);
            u4[c].h[2 * v + 1] = (_Float16)((float)sp[1] * oh.y);
        }
    }

    for (int n = 0; n < 8; ++n) {
        float bias = b_rc[n * 16 + l];
        v8f acc;
#pragma unroll
        for (int v = 0; v < 8; ++v) acc[v] = bias;
#pragma unroll
        for (int c = 0; c < 4; ++c) {
            AFrag b;
            load_bfrag(s_wrc, KP128, n * 16 + l, c, half, b);
            acc = WMMA_F16(u4[c].v, b.v, acc);
        }
#pragma unroll
        for (int v = 0; v < 8; ++v) {
            int row = e_base + v + 8 * half;
            if (row < E)
                out[(size_t)row * 128 + n * 16 + l] = gelu_fast(acc[v]);
        }
    }
}

extern "C" void kernel_launch(void* const* d_in, const int* in_sizes, int n_in,
                              void* d_out, int out_size, void* d_ws, size_t ws_size,
                              hipStream_t stream) {
    (void)n_in; (void)out_size; (void)ws_size;
    const float* known_mask   = (const float*)d_in[0];
    const int*   obs_idx      = (const int*)d_in[1];
    const int*   obs_mask_idx = (const int*)d_in[2];
    const int*   attr_idx     = (const int*)d_in[3];
    const float* obs_embs     = (const float*)d_in[4];
    const float* fea_corr     = (const float*)d_in[5];
    const float* W_rm1 = (const float*)d_in[6];
    const float* b_rm1 = (const float*)d_in[7];
    const float* W_rm2 = (const float*)d_in[8];
    const float* b_rm2 = (const float*)d_in[9];
    const float* W_rr  = (const float*)d_in[10];
    const float* b_rr  = (const float*)d_in[11];
    const float* W_rc  = (const float*)d_in[12];
    const float* b_rc  = (const float*)d_in[13];
    float* out = (float*)d_out;
    _Float16* wimg = (_Float16*)d_ws;     // 83,712 B f16 weight image

    const int E = in_sizes[1];            // obs_idx element count

    // 1) One-shot weight convert/transpose into d_ws.
    prep_weights_kernel<<<(W_TOTAL + 255) / 256, 256, 0, stream>>>(
        W_rm1, W_rm2, W_rr, W_rc, wimg);

    // 2) Main fused kernel.
    const int tiles = (E + TILE_M - 1) / TILE_M;
    const int grid  = (tiles + WAVES_PER_BLOCK - 1) / WAVES_PER_BLOCK;
    attr_relation_net_kernel<<<grid, WAVES_PER_BLOCK * 32, 0, stream>>>(
        known_mask, obs_idx, obs_mask_idx, attr_idx, obs_embs, fea_corr,
        wimg, b_rm1, b_rm2, b_rr, b_rc, out, E);
}
